// MultiHeadAttention_47820165873800
// MI455X (gfx1250) — matrix operand order repaired and checked by reading, not compiled, not run on hardware
//
#include <hip/hip_runtime.h>

// Problem constants (B,S,D,H,DK=DV fixed by the reference)
#define BATCH   4
#define S_LEN   2048
#define DMODEL  1024
#define NHEADS  16
#define HDIM    64

// 1/sqrt(64) * log2(e): folded into the Q projection so flash softmax runs in
// base-2 with zero per-score scaling VALU (exp2f -> native v_exp_f32).
#define Q_SCALE_LOG2E 0.1803368801111244f

typedef __attribute__((ext_vector_type(16))) __bf16       bf16x16;
typedef __attribute__((ext_vector_type(8)))  float        f32x8;
typedef __attribute__((ext_vector_type(4)))  float        f32x4;
typedef __attribute__((ext_vector_type(4)))  unsigned int u32x4;
typedef __attribute__((ext_vector_type(2)))  unsigned int u32x2;

union FragCvt {
  bf16x16      v;
  u32x4        q[2];
  unsigned int u[8];
};

__device__ __forceinline__ unsigned short f32_to_bf16(float f) {
  unsigned int u = __float_as_uint(f);
  u += 0x7FFFu + ((u >> 16) & 1u);   // round to nearest even
  return (unsigned short)(u >> 16);
}

__device__ __forceinline__ unsigned int pack_bf16_2(float lo, float hi) {
  return (unsigned int)f32_to_bf16(lo) | ((unsigned int)f32_to_bf16(hi) << 16);
}

// A-fragment 16x32 bf16 from row-major [M,K] (K contiguous).
// Lane r=L&15 holds row r; half=0 -> K {0..7,16..23}, half=1 -> K {8..15,24..31}.
__device__ __forceinline__ bf16x16 load_frag_a(const unsigned short* __restrict__ base,
                                               int ld, int lane) {
  const int r = lane & 15, hf = lane >> 4;
  const unsigned short* p = base + (size_t)r * ld + 8 * hf;
  FragCvt f;
  f.q[0] = *(const u32x4*)(p);
  f.q[1] = *(const u32x4*)(p + 16);
  return f.v;
}

// B-fragment 32x16 bf16 where column n of B is row n of a row-major [N,K] matrix.
// Lane n=L&15 holds column n; half=0 -> K {0..7,16..23}, half=1 -> K {8..15,24..31}.
__device__ __forceinline__ bf16x16 load_frag_b(const unsigned short* __restrict__ base,
                                               int ld, int lane) {
  const int n = lane & 15, hf = lane >> 4;
  const unsigned short* p = base + (size_t)n * ld + 8 * hf;
  FragCvt f;
  f.q[0] = *(const u32x4*)(p);
  f.q[1] = *(const u32x4*)(p + 16);
  return f.v;
}

__device__ __forceinline__ f32x8 wmma_bf16(bf16x16 a, bf16x16 b, f32x8 c) {
  return __builtin_amdgcn_wmma_f32_16x16x32_bf16(false, a, false, b, (short)0, c,
                                                 false, false);
}

#define ZERO8 ((f32x8){0.f, 0.f, 0.f, 0.f, 0.f, 0.f, 0.f, 0.f})

// ---------------------------------------------------------------- pack fp32 -> bf16
__global__ __launch_bounds__(256) void cvt_f32_bf16(const float* __restrict__ in,
                                                    unsigned short* __restrict__ out,
                                                    int n4) {
  const int stride = gridDim.x * blockDim.x;
  for (int i = blockIdx.x * blockDim.x + threadIdx.x; i < n4; i += stride) {
    f32x4 f = ((const f32x4*)in)[i];
    u32x2 o;
    o[0] = pack_bf16_2(f[0], f[1]);
    o[1] = pack_bf16_2(f[2], f[3]);
    ((u32x2*)out)[i] = o;
  }
}

// ---------------------------------------------------------------- 64x64-tile WMMA GEMM
// C[M,N] = (A[M,K](bf16) * W[N,K](bf16)^T + bias) * out_scale
// MODE 0: bf16 row-major out [M,N]
// MODE 1: bf16 per-head transposed out: Vt[b, h, d, s]   (vectorized b128 stores)
// MODE 2: f32 row-major out [M,N], no scale (final projection -> d_out)
template <int MODE>
__global__ __launch_bounds__(128) void wmma_gemm64(const unsigned short* __restrict__ A,
                                                   const unsigned short* __restrict__ W,
                                                   const float* __restrict__ bias,
                                                   void* __restrict__ Out,
                                                   int M, int N, int K,
                                                   float out_scale) {
  const int lane = threadIdx.x & 31;
  const int wid  = blockIdx.x * (blockDim.x >> 5) + (threadIdx.x >> 5);
  const int tilesN = N >> 6;
  const int m0 = (wid / tilesN) << 6;
  const int n0 = (wid % tilesN) << 6;
  if (m0 >= M) return;

  f32x8 acc[4][4];
#pragma unroll
  for (int i = 0; i < 4; ++i)
#pragma unroll
    for (int j = 0; j < 4; ++j)
      acc[i][j] = ZERO8;

  for (int k0 = 0; k0 < K; k0 += 32) {
    bf16x16 afr[4], bfr[4];
#pragma unroll
    for (int i = 0; i < 4; ++i)
      afr[i] = load_frag_a(A + (size_t)(m0 + 16 * i) * K + k0, K, lane);
#pragma unroll
    for (int j = 0; j < 4; ++j)
      bfr[j] = load_frag_b(W + (size_t)(n0 + 16 * j) * K + k0, K, lane);
#pragma unroll
    for (int i = 0; i < 4; ++i)
#pragma unroll
      for (int j = 0; j < 4; ++j)
        acc[i][j] = wmma_bf16(afr[i], bfr[j], acc[i][j]);
  }

  const int r = lane & 15, hf = lane >> 4;
#pragma unroll
  for (int j = 0; j < 4; ++j) {
    const int n = n0 + 16 * j + r;
    const float bn = bias[n];
#pragma unroll
    for (int i = 0; i < 4; ++i) {
      if (MODE == 0) {
        unsigned short* o = (unsigned short*)Out;
#pragma unroll
        for (int e = 0; e < 8; ++e) {
          const int m = m0 + 16 * i + e + 8 * hf;
          o[(size_t)m * N + n] = f32_to_bf16((acc[i][j][e] + bn) * out_scale);
        }
      } else if (MODE == 1) {
        unsigned short* o = (unsigned short*)Out;
        const int hh = n >> 6, d = n & 63;           // head, dim-in-head
        const int m  = m0 + 16 * i;                  // S_LEN % 64 == 0: tile in one batch
        const int bb = m >> 11;                      // / S_LEN
        const int s  = (m & (S_LEN - 1)) + 8 * hf;   // contiguous run of 8 in s
        u32x4 pk;
#pragma unroll
        for (int g = 0; g < 4; ++g)
          pk[g] = pack_bf16_2((acc[i][j][2 * g] + bn) * out_scale,
                              (acc[i][j][2 * g + 1] + bn) * out_scale);
        *(u32x4*)(o + (((size_t)(bb * NHEADS + hh) * HDIM + d) << 11) + s) = pk;
      } else {
        float* o = (float*)Out;
#pragma unroll
        for (int e = 0; e < 8; ++e) {
          const int m = m0 + 16 * i + e + 8 * hf;
          o[(size_t)m * N + n] = acc[i][j][e] + bn;
        }
      }
    }
  }
}

// ---------------------------------------------------------------- flash attention
// One wave per (b, h, 32-query tile): two 16-query subtiles share every K
// A-fragment and V B-fragment (2x arithmetic intensity vs 16-query waves).
// Computes S^T = K*Q^T so each lane owns one query column per subtile: softmax
// reductions are in-lane (+ one shfl_xor 16), and the two 16-key S^T C-frags
// repack per-lane (no data movement) into the 16x32 P A-fragment.
// Q was pre-scaled by (1/sqrt(dk))*log2(e): softmax runs in base 2 via exp2f.
__global__ __launch_bounds__(128) void flash_attn_wmma(const unsigned short* __restrict__ Qh,
                                                       const unsigned short* __restrict__ Kh,
                                                       const unsigned short* __restrict__ Vt,
                                                       unsigned short* __restrict__ Oa) {
  const int lane = threadIdx.x & 31;
  const int wid  = blockIdx.x * (blockDim.x >> 5) + (threadIdx.x >> 5);
  const int qt = wid & (S_LEN / 32 - 1);          // 64 query tiles of 32
  const int h  = (wid >> 6) & (NHEADS - 1);
  const int b  = wid >> 10;
  const int q0 = qt << 5;
  const int hf = lane >> 4;

  const unsigned short* Qbase = Qh + ((size_t)(b * S_LEN + q0) * NHEADS + h) * HDIM;
  const unsigned short* Kbase = Kh + ((size_t)(b * S_LEN) * NHEADS + h) * HDIM;
  const unsigned short* Vbase = Vt + ((size_t)(b * NHEADS + h) * HDIM) * S_LEN;

  // Q^T B-fragments, hoisted for the whole key loop: [subtile][dstep]
  bf16x16 bq[2][2];
#pragma unroll
  for (int qs = 0; qs < 2; ++qs) {
    const unsigned short* qp = Qbase + (size_t)(16 * qs) * DMODEL;
    bq[qs][0] = load_frag_b(qp, DMODEL, lane);
    bq[qs][1] = load_frag_b(qp + 32, DMODEL, lane);
  }

  f32x8 o[2][4];
#pragma unroll
  for (int qs = 0; qs < 2; ++qs)
#pragma unroll
    for (int dt = 0; dt < 4; ++dt) o[qs][dt] = ZERO8;
  float m_run[2] = {-3.4e38f, -3.4e38f};
  float l_run[2] = {0.f, 0.f};

  for (int j0 = 0; j0 < S_LEN; j0 += 32) {
    // K A-fragments for 32 keys, shared by both query subtiles: [jt][dstep]
    bf16x16 kfr[2][2];
#pragma unroll
    for (int jt = 0; jt < 2; ++jt) {
      const unsigned short* kp = Kbase + (size_t)(j0 + 16 * jt) * DMODEL;
      kfr[jt][0] = load_frag_a(kp, DMODEL, lane);
      kfr[jt][1] = load_frag_a(kp + 32, DMODEL, lane);
    }

    FragCvt p[2];       // P as 16x32 A-fragments, one per query subtile
    float   af[2][8];   // per-row rescale factors for O accumulators
#pragma unroll
    for (int qs = 0; qs < 2; ++qs) {
      f32x8 st0 = wmma_bf16(kfr[0][0], bq[qs][0], ZERO8);
      st0       = wmma_bf16(kfr[0][1], bq[qs][1], st0);
      f32x8 st1 = wmma_bf16(kfr[1][0], bq[qs][0], ZERO8);
      st1       = wmma_bf16(kfr[1][1], bq[qs][1], st1);

      // lane owns query q = lane&15 of this subtile; 16 log2-domain scores
      float mloc = -3.4e38f;
#pragma unroll
      for (int e = 0; e < 8; ++e)
        mloc = fmaxf(mloc, fmaxf(st0[e], st1[e]));
      mloc = fmaxf(mloc, __shfl_xor(mloc, 16, 32));    // merge key halves
      const float m_new = fmaxf(m_run[qs], mloc);
      const float alpha = exp2f(m_run[qs] - m_new);

      float lloc = 0.f;
      float pe0[8], pe1[8];
#pragma unroll
      for (int e = 0; e < 8; ++e) {
        pe0[e] = exp2f(st0[e] - m_new);
        pe1[e] = exp2f(st1[e] - m_new);
        lloc += pe0[e] + pe1[e];
      }
#pragma unroll
      for (int g = 0; g < 4; ++g) {
        p[qs].u[g]     = pack_bf16_2(pe0[2 * g], pe0[2 * g + 1]);
        p[qs].u[g + 4] = pack_bf16_2(pe1[2 * g], pe1[2 * g + 1]);
      }
      lloc += __shfl_xor(lloc, 16, 32);
      l_run[qs] = l_run[qs] * alpha + lloc;
      m_run[qs] = m_new;

      // O accum has q along VGPR index: fetch per-row alpha from owning lane
#pragma unroll
      for (int e = 0; e < 8; ++e) af[qs][e] = __shfl(alpha, e + 8 * hf, 32);
#pragma unroll
      for (int dt = 0; dt < 4; ++dt) {
#pragma unroll
        for (int e = 0; e < 8; ++e) o[qs][dt][e] *= af[qs][e];
      }
    }

    // O += P * V : each V B-fragment (contiguous b128 loads) feeds both subtiles
#pragma unroll
    for (int dt = 0; dt < 4; ++dt) {
      bf16x16 bv = load_frag_b(Vbase + ((size_t)(16 * dt) << 11) + j0, S_LEN, lane);
      o[0][dt] = wmma_bf16(p[0].v, bv, o[0][dt]);
      o[1][dt] = wmma_bf16(p[1].v, bv, o[1][dt]);
    }
  }

  // normalize and store O as bf16 in [B, S, H*DV] for the output projection
  const int r = lane & 15;
#pragma unroll
  for (int qs = 0; qs < 2; ++qs) {
    float linv[8];
#pragma unroll
    for (int e = 0; e < 8; ++e) linv[e] = 1.f / __shfl(l_run[qs], e + 8 * hf, 32);
#pragma unroll
    for (int dt = 0; dt < 4; ++dt) {
#pragma unroll
      for (int e = 0; e < 8; ++e) {
        const int q = q0 + 16 * qs + e + 8 * hf;
        const int d = 16 * dt + r;
        Oa[((size_t)(b * S_LEN + q) * NHEADS + h) * HDIM + d] =
            f32_to_bf16(o[qs][dt][e] * linv[e]);
      }
    }
  }
}

// ---------------------------------------------------------------- host launch
extern "C" void kernel_launch(void* const* d_in, const int* in_sizes, int n_in,
                              void* d_out, int out_size, void* d_ws, size_t ws_size,
                              hipStream_t stream) {
  (void)in_sizes; (void)n_in; (void)out_size; (void)ws_size;

  const float* q  = (const float*)d_in[0];
  const float* k  = (const float*)d_in[1];
  const float* v  = (const float*)d_in[2];
  const float* wq = (const float*)d_in[3];
  const float* bq = (const float*)d_in[4];
  const float* wk = (const float*)d_in[5];
  const float* bk = (const float*)d_in[6];
  const float* wv = (const float*)d_in[7];
  const float* bv = (const float*)d_in[8];
  const float* wo = (const float*)d_in[9];
  const float* bo = (const float*)d_in[10];

  const size_t NTOK = (size_t)BATCH * S_LEN;    // 8192
  const size_t SZ_X = NTOK * DMODEL;            // activations, elements
  const size_t SZ_W = (size_t)DMODEL * DMODEL;  // weights, elements

  unsigned short* qb  = (unsigned short*)d_ws;  // bf16 staging (ws: ~120 MB used)
  unsigned short* kb  = qb  + SZ_X;
  unsigned short* vb  = kb  + SZ_X;
  unsigned short* wqb = vb  + SZ_X;
  unsigned short* wkb = wqb + SZ_W;
  unsigned short* wvb = wkb + SZ_W;
  unsigned short* wob = wvb + SZ_W;
  unsigned short* Qp  = wob + SZ_W;             // [B,S,H,DK], pre-scaled by log2e/sqrt(dk)
  unsigned short* Kp  = Qp  + SZ_X;             // [B,S,H,DK]
  unsigned short* Vtp = Kp  + SZ_X;             // [B,H,DV,S]
  unsigned short* att = Vtp + SZ_X;             // [B,S,H*DV]

  cvt_f32_bf16<<<2048, 256, 0, stream>>>(q,  qb,  (int)(SZ_X >> 2));
  cvt_f32_bf16<<<2048, 256, 0, stream>>>(k,  kb,  (int)(SZ_X >> 2));
  cvt_f32_bf16<<<2048, 256, 0, stream>>>(v,  vb,  (int)(SZ_X >> 2));
  cvt_f32_bf16<<<512,  256, 0, stream>>>(wq, wqb, (int)(SZ_W >> 2));
  cvt_f32_bf16<<<512,  256, 0, stream>>>(wk, wkb, (int)(SZ_W >> 2));
  cvt_f32_bf16<<<512,  256, 0, stream>>>(wv, wvb, (int)(SZ_W >> 2));
  cvt_f32_bf16<<<512,  256, 0, stream>>>(wo, wob, (int)(SZ_W >> 2));

  const int gemmBlocks = (int)(((NTOK >> 6) * (DMODEL >> 6)) >> 2);  // 512 (4 waves/blk)
  wmma_gemm64<0><<<gemmBlocks, 128, 0, stream>>>(qb, wqb, bq, Qp,  (int)NTOK, DMODEL,
                                                 DMODEL, Q_SCALE_LOG2E);
  wmma_gemm64<0><<<gemmBlocks, 128, 0, stream>>>(kb, wkb, bk, Kp,  (int)NTOK, DMODEL,
                                                 DMODEL, 1.0f);
  wmma_gemm64<1><<<gemmBlocks, 128, 0, stream>>>(vb, wvb, bv, Vtp, (int)NTOK, DMODEL,
                                                 DMODEL, 1.0f);

  // B*H*(S/32) = 4096 waves, 4 waves per block
  flash_attn_wmma<<<1024, 128, 0, stream>>>(Qp, Kp, Vtp, att);

  wmma_gemm64<2><<<gemmBlocks, 128, 0, stream>>>(att, wob, bo, d_out,
                                                 (int)NTOK, DMODEL, DMODEL, 1.0f);
}
